// attn_2_vote_5970004541820
// MI455X (gfx1250) — compile-verified, matrix operand order checked by reading, not run on hardware
//
#include <hip/hip_runtime.h>

// ---------------------------------------------------------------------------
// CDNA5 (gfx1250, wave32) implementation.
// GEMMs: one 16x16 C tile per wave via v_wmma_f32_16x16x32_bf16 (f32 accum),
// A tile staged global->LDS with CDNA5 async loads (ASYNCcnt path).
// ---------------------------------------------------------------------------

typedef __attribute__((ext_vector_type(16))) __bf16 v16bf;
typedef __attribute__((ext_vector_type(8)))  float  v8f;

namespace {
constexpr int kB       = 2;
constexpr int kN       = 16384;
constexpr int kC       = 128;               // K dim of every GEMM (fixed)
constexpr int kVF      = 3;
constexpr int kV       = kN * kVF;          // 49152 votes per sample
constexpr int kCo3     = (3 + kC) * kVF;    // 393
constexpr int kCo3P    = 400;               // padded to 25 16-col tiles
constexpr int kP       = 4;
constexpr int kHash    = 1 << 17;           // open-addressing table (> 2*V)
constexpr int kHashMask= kHash - 1;
constexpr int kRowTiles= kN / 16;           // 1024
constexpr float kEps   = 1e-5f;
constexpr float kInvGrid = 10.0f;           // 1 / GRID
} // namespace

// Pack 16 f32 (two groups of 8 consecutive K values) into a bf16 WMMA operand.
__device__ inline v16bf pack16(float4 a, float4 b, float4 c, float4 d) {
  v16bf r;
  r[0]=(__bf16)a.x;  r[1]=(__bf16)a.y;  r[2]=(__bf16)a.z;  r[3]=(__bf16)a.w;
  r[4]=(__bf16)b.x;  r[5]=(__bf16)b.y;  r[6]=(__bf16)b.z;  r[7]=(__bf16)b.w;
  r[8]=(__bf16)c.x;  r[9]=(__bf16)c.y;  r[10]=(__bf16)c.z; r[11]=(__bf16)c.w;
  r[12]=(__bf16)d.x; r[13]=(__bf16)d.y; r[14]=(__bf16)d.z; r[15]=(__bf16)d.w;
  return r;
}

// Zero-pad W3 (393x128 -> 400x128) and b3 (393 -> 400) so the GEMM inner loop
// is branch-free for every layer.
__global__ void pad_w3(const float* __restrict__ W3, const float* __restrict__ b3,
                       float* __restrict__ WP, float* __restrict__ BP) {
  int i = blockIdx.x * blockDim.x + threadIdx.x;
  if (i < kCo3P * kC) {
    int r = i / kC;
    WP[i] = (r < kCo3) ? W3[i] : 0.0f;   // same row-major index while r < 393
  }
  if (i < kCo3P) BP[i] = (i < kCo3) ? b3[i] : 0.0f;
}

// Y[b,n,co] = X[b,n,:] . W[co,:] + bias[co]  (+ deterministic BN partials).
// One wave -> one 16x16 output tile; CoPad is a multiple of 16 and W/bias are
// zero-padded to CoPad rows. A layout (16-bit, 16x32): lanes 0-15 hold row
// M=lane with K groups {k..k+7, k+16..k+23}; lanes 16-31 hold {k+8.., k+24..}.
__global__ __launch_bounds__(256, 2)
void gemm_bias_wmma(const float* __restrict__ X, const float* __restrict__ W,
                    const float* __restrict__ bias, float* __restrict__ Y,
                    float* __restrict__ partial, int N, int CoPad) {
  __shared__ __align__(16) float atile[16][132];    // 16x128 A tile, stride 132
  __shared__ __align__(16) float tile[8][16][20];   // epilogue transpose
  const int tid  = threadIdx.x;
  const int wave = tid >> 5;
  const int lane = tid & 31;
  const int hl   = lane >> 4;        // K-half selector
  const int lr   = lane & 15;        // row (A) / col (B) within tile
  const int rowTile = blockIdx.x;
  const int b    = blockIdx.z;
  const int row0 = rowTile << 4;
  const int nColTiles = CoPad >> 4;
  int colTile = wave + (blockIdx.y << 3);
  const bool active = colTile < nColTiles;
  if (!active) colTile = nColTiles - 1;    // keep wave alive for barriers
  const int col0 = colTile << 4;
  const int col  = col0 + lr;

  // --- Stage the shared 16x128 A tile into LDS with CDNA5 async loads.
  // 512 x 16B segments; each of the 256 threads issues two async transfers.
  {
    const float* __restrict__ xbase = X + ((size_t)b * N + row0) * kC;
#pragma unroll
    for (int j = 0; j < 2; ++j) {
      int li  = j * 256 + tid;            // 0..511
      int row = li >> 5;                  // 0..15
      int seg = li & 31;                  // 0..31 (16B each)
      const float* g = xbase + (size_t)row * kC + seg * 4;
      unsigned lds = (unsigned)(size_t)&atile[row][seg * 4];
      asm volatile("global_load_async_to_lds_b128 %0, %1, off"
                   :: "v"(lds), "v"(g) : "memory");
    }
    asm volatile("s_wait_asynccnt 0x0" ::: "memory");
  }
  __syncthreads();

  const float* __restrict__ wrow = W + (size_t)col * kC;
  v8f acc = {};
#pragma unroll
  for (int s = 0; s < kC / 32; ++s) {     // fully unrolled: 4 WMMAs
    const int k0 = s * 32 + (hl << 3);
    float4 a0 = *(const float4*)&atile[lr][k0];
    float4 a1 = *(const float4*)&atile[lr][k0 + 4];
    float4 a2 = *(const float4*)&atile[lr][k0 + 16];
    float4 a3 = *(const float4*)&atile[lr][k0 + 20];
    float4 w0 = *(const float4*)(wrow + k0);
    float4 w1 = *(const float4*)(wrow + k0 + 4);
    float4 w2 = *(const float4*)(wrow + k0 + 16);
    float4 w3 = *(const float4*)(wrow + k0 + 20);
    v16bf A  = pack16(a0, a1, a2, a3);
    v16bf Bm = pack16(w0, w1, w2, w3);
    acc = __builtin_amdgcn_wmma_f32_16x16x32_bf16(false, A, false, Bm,
                                                  (short)0, acc, false, false);
  }

  // Epilogue: bias, deterministic BN partial sums, LDS transpose, b128 stores.
  const float bc = bias[col];
  float s1 = 0.0f, s2 = 0.0f;
#pragma unroll
  for (int j = 0; j < 8; ++j) {           // C/D layout: VGPR j -> M = j + 8*hl
    float v = acc[j] + bc;
    s1 += v; s2 += v * v;
    tile[wave][j + (hl << 3)][lr] = v;
  }
  if (active && partial != nullptr) {
    s1 += __shfl_xor(s1, 16);             // combine row-halves per column
    s2 += __shfl_xor(s2, 16);
    if (hl == 0) {
      size_t pb = (((size_t)b * kRowTiles + rowTile) * 2) * CoPad + col;
      partial[pb]         = s1;
      partial[pb + CoPad] = s2;
    }
  }
  const int srow = lane >> 1;
  const int c0   = (lane & 1) << 3;
  float4 o0 = *(const float4*)&tile[wave][srow][c0];
  float4 o1 = *(const float4*)&tile[wave][srow][c0 + 4];
  if (active) {
    float* yp = Y + ((size_t)b * N + row0 + srow) * CoPad + col0 + c0;
    *(float4*)yp       = o0;
    *(float4*)(yp + 4) = o1;
  }
}

// Deterministic reduction of per-rowtile partials -> stats[b][2][C].
__global__ void stats_reduce(const float* __restrict__ partial,
                             float* __restrict__ stats, int tiles, int C) {
  int i = blockIdx.x * blockDim.x + threadIdx.x;
  if (i >= kB * 2 * C) return;
  int c = i % C;
  int s = (i / C) & 1;
  int b = i / (2 * C);
  float acc = 0.0f;
  for (int t = 0; t < tiles; ++t)
    acc += partial[(((size_t)b * tiles + t) * 2 + s) * C + c];
  stats[i] = acc;
}

__global__ void bn_relu_kernel(float* __restrict__ Y, const float* __restrict__ stats,
                               const float* __restrict__ g, const float* __restrict__ be) {
  size_t i = (size_t)blockIdx.x * blockDim.x + threadIdx.x;
  constexpr size_t total = (size_t)kB * kN * kC;
  if (i >= total) return;
  int c = (int)(i % kC);
  int b = (int)(i / ((size_t)kN * kC));
  float m  = stats[(b * 2 + 0) * kC + c] * (1.0f / kN);
  float v  = stats[(b * 2 + 1) * kC + c] * (1.0f / kN) - m * m;
  float x  = Y[i];
  float y  = (x - m) * __frsqrt_rn(v + kEps) * g[c] + be[c];
  Y[i] = fmaxf(y, 0.0f);
}

// Per-vote: offsets -> coords, grid indices, 30-bit voxel key.
__global__ void voxelize_kernel(const int* __restrict__ ID, const float* __restrict__ CO,
                                const float* __restrict__ LOGb, int* __restrict__ INDv,
                                float* __restrict__ COORv, int* __restrict__ KEYv) {
  int v = blockIdx.x * blockDim.x + threadIdx.x;
  int b = blockIdx.y;
  if (v >= kV) return;
  int n = v / kVF, f = v % kVF;
  const float* lrow = LOGb + ((size_t)b * kN + n) * kCo3P + f * (3 + kC);
  size_t nb = (size_t)b * kN + n;
  size_t vb = (size_t)b * kV + v;
  int key = 0;
#pragma unroll
  for (int d = 0; d < 3; ++d) {
    float off  = lrow[d];
    float coor = CO[nb * 3 + d] + off;
    int gr  = (int)(off * kInvGrid);     // trunc-toward-zero == astype(int32)
    int ind = ID[nb * 3 + d] + gr;
    INDv[vb * 3 + d]  = ind;
    COORv[vb * 3 + d] = coor;
    int hc = ind < -512 ? -512 : (ind > 511 ? 511 : ind);
    key = key * 1024 + (hc + 512);
  }
  KEYv[vb] = key;
}

__global__ void hash_reset(int* __restrict__ Tkey, int* __restrict__ Tcnt,
                           int* __restrict__ Tm) {
  int i = blockIdx.x * blockDim.x + threadIdx.x;
  if (i >= kB * kHash) return;
  const size_t S = (size_t)kB * kHash;
  Tkey[i] = -1;
  Tcnt[i] = 0;
  Tm[i]         = 0x7fffffff;
  Tm[i + S]     = 0x7fffffff;
  Tm[i + 2 * S] = 0x7fffffff;
  Tm[i + 3 * S] = 0x7fffffff;
}

__global__ void first_reset(int* __restrict__ firstSlot) {
  int i = blockIdx.x * blockDim.x + threadIdx.x;
  if (i < kB * kV) firstSlot[i] = -1;
}

// Insert keys; per-key count + minimum member index. Physical slot assignment
// is timing-dependent but all OUTPUTS depend only on per-key content -> stable.
__global__ void hash_insert(const int* __restrict__ KEYv, int* __restrict__ Tkey,
                            int* __restrict__ Tcnt, int* __restrict__ Tm0,
                            int* __restrict__ SLOTv) {
  int v = blockIdx.x * blockDim.x + threadIdx.x;
  int b = blockIdx.y;
  if (v >= kV) return;
  int key = KEYv[(size_t)b * kV + v];
  int* tk = Tkey + (size_t)b * kHash;
  unsigned h = ((unsigned)key * 2654435761u) >> 8;
  h &= kHashMask;
  while (true) {
    int old = atomicCAS(&tk[h], -1, key);
    if (old == -1 || old == key) break;
    h = (h + 1) & kHashMask;
  }
  size_t idx = (size_t)b * kHash + h;
  atomicAdd(&Tcnt[idx], 1);
  atomicMin(&Tm0[idx], v);
  SLOTv[(size_t)b * kV + v] = (int)h;
}

// Rank pass: deterministically find the stage-th smallest member per key.
__global__ void hash_members(const int* __restrict__ SLOTv, const int* __restrict__ Tm,
                             int* __restrict__ Tout, int stage) {
  int v = blockIdx.x * blockDim.x + threadIdx.x;
  int b = blockIdx.y;
  if (v >= kV) return;
  const size_t S = (size_t)kB * kHash;
  size_t idx = (size_t)b * kHash + SLOTv[(size_t)b * kV + v];
  if (v == Tm[idx]) return;
  if (stage >= 2 && v == Tm[idx + S]) return;
  if (stage >= 3 && v == Tm[idx + 2 * S]) return;
  atomicMin(&Tout[idx], v);
}

__global__ void mark_first(const int* __restrict__ SLOTv, const int* __restrict__ Tm0,
                           int* __restrict__ firstSlot) {
  int v = blockIdx.x * blockDim.x + threadIdx.x;
  int b = blockIdx.y;
  if (v >= kV) return;
  size_t vb = (size_t)b * kV + v;
  int slot = SLOTv[vb];
  if (Tm0[(size_t)b * kHash + slot] == v) firstSlot[vb] = slot;
}

// Final gather: each block handles one output slot u (placed at minIndex of
// its key). Writes EVERY output element (zeros for empty slots).
__global__ __launch_bounds__(128)
void scatter_out(const int* __restrict__ firstSlot, const int* __restrict__ Tcnt,
                 const int* __restrict__ Tm, const int* __restrict__ INDv,
                 const float* __restrict__ COORv, const float* __restrict__ FT,
                 const float* __restrict__ LOGb, int* __restrict__ outInd,
                 float* __restrict__ outCoor, float* __restrict__ outFeat,
                 float* __restrict__ outMask) {
  const int u = blockIdx.x;
  const int b = blockIdx.y;
  const int t = threadIdx.x;                 // 0..127 = channel
  const size_t ub = (size_t)b * kV + u;
  const int slot = firstSlot[ub];
  if (slot < 0) {
    if (t < 3)       outInd[ub * 3 + t] = 0;
    if (t == 4)      outMask[ub] = 0.0f;
    if (t < kP * 3)  outCoor[ub * kP * 3 + t] = 0.0f;
#pragma unroll
    for (int p = 0; p < kP; ++p) outFeat[(ub * kP + p) * kC + t] = 0.0f;
    return;
  }
  const size_t S   = (size_t)kB * kHash;
  const size_t idx = (size_t)b * kHash + slot;
  const int cnt = Tcnt[idx];
  int m[4];
  m[0] = u;                 // == Tm0 (this slot IS the min index)
  m[1] = Tm[idx + S];
  m[2] = Tm[idx + 2 * S];
  m[3] = Tm[idx + 3 * S];
  const int eff = cnt < kP ? cnt : kP;       // p % count == p % min(count,P), p<P
  if (t < 3)      outInd[ub * 3 + t] = INDv[ub * 3 + t];
  if (t == 4)     outMask[ub] = 1.0f;        // counts*RATIO_MAX >= P for cnt>=1
  if (t < kP * 3) {
    int p = t / 3, d = t % 3;
    int s = m[p % eff];
    outCoor[ub * kP * 3 + t] = COORv[((size_t)b * kV + s) * 3 + d];
  }
#pragma unroll
  for (int p = 0; p < kP; ++p) {
    int s = m[p % eff];
    int n = s / kVF;
    size_t nb = (size_t)b * kN + n;
    // feat = FT[n] + log[n, VF-1, 3:] * (3/VF == 1.0)
    float fo = LOGb[nb * kCo3P + (kVF - 1) * (3 + kC) + 3 + t];
    outFeat[(ub * kP + p) * kC + t] = FT[nb * kC + t] + fo;
  }
}

extern "C" void kernel_launch(void* const* d_in, const int* in_sizes, int n_in,
                              void* d_out, int out_size, void* d_ws, size_t ws_size,
                              hipStream_t stream) {
  (void)in_sizes; (void)n_in; (void)out_size; (void)ws_size;
  const int*   ID  = (const int*)  d_in[0];
  const float* CO  = (const float*)d_in[1];
  const float* FT  = (const float*)d_in[2];
  const float* W1  = (const float*)d_in[3];
  const float* b1  = (const float*)d_in[4];
  const float* g1  = (const float*)d_in[5];
  const float* be1 = (const float*)d_in[6];
  const float* W2  = (const float*)d_in[7];
  const float* b2  = (const float*)d_in[8];
  const float* g2  = (const float*)d_in[9];
  const float* be2 = (const float*)d_in[10];
  const float* W3  = (const float*)d_in[11];
  const float* b3  = (const float*)d_in[12];

  char* ws = (char*)d_ws;
  size_t off = 0;
  auto take = [&](size_t bytes) -> char* {
    char* p = ws + off;
    off += (bytes + 255) & ~(size_t)255;
    return p;
  };
  float* H1    = (float*)take((size_t)kB * kN * kC * 4);
  float* H2    = (float*)take((size_t)kB * kN * kC * 4);
  float* LOGb  = (float*)take((size_t)kB * kN * kCo3P * 4);
  float* PART  = (float*)take((size_t)kB * kRowTiles * 2 * kC * 4);
  float* STATS = (float*)take((size_t)kB * 2 * kC * 4);
  float* WP    = (float*)take((size_t)kCo3P * kC * 4);
  float* BP    = (float*)take((size_t)kCo3P * 4);
  int*   INDv  = (int*)  take((size_t)kB * kV * 3 * 4);
  float* COORv = (float*)take((size_t)kB * kV * 3 * 4);
  int*   KEYv  = (int*)  take((size_t)kB * kV * 4);
  int*   SLOTv = (int*)  take((size_t)kB * kV * 4);
  int*   FIRST = (int*)  take((size_t)kB * kV * 4);
  int*   TKEY  = (int*)  take((size_t)kB * kHash * 4);
  int*   TCNT  = (int*)  take((size_t)kB * kHash * 4);
  int*   TM    = (int*)  take((size_t)4 * kB * kHash * 4);

  float* fout    = (float*)d_out;
  int*   outInd  = (int*)fout;
  float* outCoor = fout + (size_t)kB * kV * 3;
  float* outFeat = outCoor + (size_t)kB * kV * kP * 3;
  float* outMask = outFeat + (size_t)kB * kV * kP * kC;

  const dim3 blk(256);
  const dim3 gGemm(kRowTiles, 1, kB);
  const dim3 gGemm3(kRowTiles, 4, kB);      // 25 col tiles / 8 waves -> 4 y-blocks
  const dim3 gStats((kB * 2 * kC + 255) / 256);
  const dim3 gBN((unsigned)(((size_t)kB * kN * kC) / 256));
  const dim3 gV((kV + 255) / 256, kB);
  const dim3 gHashR((kB * kHash + 255) / 256);
  const dim3 gFirstR((kB * kV + 255) / 256);

  // Pad W3/b3 once so every GEMM runs a branch-free inner loop.
  pad_w3<<<dim3((kCo3P * kC + 255) / 256), blk, 0, stream>>>(W3, b3, WP, BP);

  // Layer 1: FT @ W1^T + b1 -> BN stats -> BN+ReLU
  gemm_bias_wmma<<<gGemm, blk, 0, stream>>>(FT, W1, b1, H1, PART, kN, kC);
  stats_reduce  <<<gStats, blk, 0, stream>>>(PART, STATS, kRowTiles, kC);
  bn_relu_kernel<<<gBN, blk, 0, stream>>>(H1, STATS, g1, be1);
  // Layer 2
  gemm_bias_wmma<<<gGemm, blk, 0, stream>>>(H1, W2, b2, H2, PART, kN, kC);
  stats_reduce  <<<gStats, blk, 0, stream>>>(PART, STATS, kRowTiles, kC);
  bn_relu_kernel<<<gBN, blk, 0, stream>>>(H2, STATS, g2, be2);
  // Layer 3: logits (393 cols, zero-padded weights to 400)
  gemm_bias_wmma<<<gGemm3, blk, 0, stream>>>(H2, WP, BP, LOGb, nullptr, kN, kCo3P);

  // Voxelize + dedup + gather
  voxelize_kernel<<<gV, blk, 0, stream>>>(ID, CO, LOGb, INDv, COORv, KEYv);
  hash_reset <<<gHashR, blk, 0, stream>>>(TKEY, TCNT, TM);
  first_reset<<<gFirstR, blk, 0, stream>>>(FIRST);
  hash_insert<<<gV, blk, 0, stream>>>(KEYv, TKEY, TCNT, TM, SLOTv);
  const size_t S = (size_t)kB * kHash;
  hash_members<<<gV, blk, 0, stream>>>(SLOTv, TM, TM + S, 1);
  hash_members<<<gV, blk, 0, stream>>>(SLOTv, TM, TM + 2 * S, 2);
  hash_members<<<gV, blk, 0, stream>>>(SLOTv, TM, TM + 3 * S, 3);
  mark_first  <<<gV, blk, 0, stream>>>(SLOTv, TM, FIRST);
  scatter_out<<<dim3(kV, kB), dim3(kC), 0, stream>>>(FIRST, TCNT, TM, INDv, COORv,
                                                     FT, LOGb, outInd, outCoor,
                                                     outFeat, outMask);
}
